// AgentAttention_4355096838971
// MI455X (gfx1250) — matrix operand order repaired and checked by reading, compiled, not run on hardware
//
#include <hip/hip_runtime.h>

typedef __bf16 bf16_t;
typedef __attribute__((ext_vector_type(16))) __bf16 v16bf;
typedef __attribute__((ext_vector_type(8)))  __bf16 v8bf;
typedef __attribute__((ext_vector_type(4)))  __bf16 v4bf;
typedef __attribute__((ext_vector_type(8)))  float  v8f;
typedef __attribute__((ext_vector_type(4)))  float  v4f;

// Problem constants
#define NHEAD 12
#define NAGENT 16
#define CH 768
#define HDIM 64
#define NB 8
#define NTOK 4096
#define MTOT (NB * NTOK)   // 32768
#define BH (NB * NHEAD)    // 96

static __device__ __forceinline__ v8f wmma_bf16f32(v16bf a, v16bf b, v8f c) {
  // D = A(16xK) * B(Kx16) + C, K=32, fp32 accumulate
  return __builtin_amdgcn_wmma_f32_16x16x32_bf16(false, a, false, b, (short)0, c, false, false);
}

// A-fragment loader. Source is row-major [row][K] (pitch in elements, mult of 8).
// ISA layout (16-bit A 16x32): lane<16 -> row=lane, K in {ko..ko+7, 16+ko..16+ko+7}, ko = (lane/16)*8.
static __device__ __forceinline__ v16bf load_fragA(const bf16_t* base, int row, int pitch,
                                                   int k0, int lane) {
  int r  = row + (lane & 15);
  int ko = k0 + ((lane >> 4) << 3);
  v8bf lo = *(const v8bf*)(base + (size_t)r * pitch + ko);
  v8bf hi = *(const v8bf*)(base + (size_t)r * pitch + ko + 16);
  return __builtin_shufflevector(lo, hi, 0, 1, 2, 3, 4, 5, 6, 7, 8, 9, 10, 11, 12, 13, 14, 15);
}

// B-fragment loader. Source is B^T stored row-major [N][K] (pitch mult of 8).
// ISA layout (16-bit B 32x16): lane<16 -> col n=lane, K in {ko..ko+7, 16+ko..16+ko+7}, ko = (lane/16)*8.
static __device__ __forceinline__ v16bf load_fragB(const bf16_t* base, int row, int pitch,
                                                   int k0, int lane) {
  int r  = row + (lane & 15);
  int ko = k0 + ((lane >> 4) << 3);
  v8bf lo = *(const v8bf*)(base + (size_t)r * pitch + ko);
  v8bf hi = *(const v8bf*)(base + (size_t)r * pitch + ko + 16);
  return __builtin_shufflevector(lo, hi, 0, 1, 2, 3, 4, 5, 6, 7, 8, 9, 10, 11, 12, 13, 14, 15);
}

// ---------------------------------------------------------------------------
// Kernel 1: ab[h][a] bias (stage-2). pb (stage-1) is constant along the
// softmax axis => provable no-op, skipped.
// ab[h,a] = mean(bilinear_resize_16x16(na_bias[h,a,7,7]))
//         + mean_u ha_bias[h,u,a] + mean_u wa_bias[h,u,a]
// ---------------------------------------------------------------------------
__global__ void bias_kernel(const float* __restrict__ na_bias,
                            const float* __restrict__ ha_bias,
                            const float* __restrict__ wa_bias,
                            float* __restrict__ ab) {
  int t = threadIdx.x;
  if (t >= NHEAD * NAGENT) return;
  int h = t >> 4, a = t & 15;
  const float* img = na_bias + (size_t)t * 49;
  float s = 0.f;
  for (int i = 0; i < 16; ++i) {
    float fy = (i + 0.5f) * 0.4375f - 0.5f;           // (i+0.5)*7/16 - 0.5, half-pixel
    fy = fminf(fmaxf(fy, 0.f), 6.f);
    int y0 = (int)fy; if (y0 > 5) y0 = 5;
    float wy = fy - (float)y0;
    for (int j = 0; j < 16; ++j) {
      float fx = (j + 0.5f) * 0.4375f - 0.5f;
      fx = fminf(fmaxf(fx, 0.f), 6.f);
      int x0 = (int)fx; if (x0 > 5) x0 = 5;
      float wx = fx - (float)x0;
      float v00 = img[y0 * 7 + x0],     v01 = img[y0 * 7 + x0 + 1];
      float v10 = img[(y0 + 1) * 7 + x0], v11 = img[(y0 + 1) * 7 + x0 + 1];
      s += v00 * (1.f - wy) * (1.f - wx) + v01 * (1.f - wy) * wx
         + v10 * wy * (1.f - wx) + v11 * wy * wx;
    }
  }
  s *= (1.f / 256.f);
  float sh = 0.f, sw = 0.f;
  for (int u = 0; u < 16; ++u) {
    sh += ha_bias[((size_t)h * 16 + u) * 16 + a];   // (1,H,A,1,A)
    sw += wa_bias[((size_t)h * 16 + u) * 16 + a];   // (1,H,1,A,A)
  }
  ab[t] = s + (sh + sw) * (1.f / 16.f);
}

// ---------------------------------------------------------------------------
// Kernel 2: fused Q/K/V projection GEMM, fp32 in -> bf16 staged in LDS ->
// v_wmma_f32_16x16x32_bf16.  128x128 tile / block, 8 waves in 2(M)x4(N) grid,
// each wave 64x32 (8 accumulator tiles).  V is written channel-transposed
// (Vt[b][c][tok]) straight out of the D fragment layout.
// ---------------------------------------------------------------------------
__global__ __launch_bounds__(256) void qkv_gemm_kernel(
    const float* __restrict__ s1, const float* __restrict__ s2,
    const float* __restrict__ Wq, const float* __restrict__ Wk, const float* __restrict__ Wv,
    const float* __restrict__ bq, const float* __restrict__ bk, const float* __restrict__ bv,
    bf16_t* __restrict__ Qbf, bf16_t* __restrict__ Kbf, bf16_t* __restrict__ Vt) {
  const int which = blockIdx.z;
  const float* X    = (which == 0) ? s1 : s2;
  const float* W    = (which == 0) ? Wq : (which == 1 ? Wk : Wv);
  const float* bias = (which == 0) ? bq : (which == 1 ? bk : bv);
  const int mBase = blockIdx.x * 128;
  const int nBase = blockIdx.y * 128;
  __shared__ bf16_t As[128 * 40];   // pitch 40 elems (80B) -> conflict-free b128 reads
  __shared__ bf16_t Bs[128 * 40];
  const int tid  = threadIdx.x;
  const int lane = tid & 31;
  const int wave = tid >> 5;
  const int wm = wave & 1;    // 0..1 : 64-row group
  const int wn = wave >> 1;   // 0..3 : 32-col group
  v8f acc[4][2];
  for (int i = 0; i < 4; ++i)
    for (int j = 0; j < 2; ++j)
      for (int e = 0; e < 8; ++e) acc[i][j][e] = 0.f;

  for (int k0 = 0; k0 < CH; k0 += 32) {
    __syncthreads();
    for (int it = 0; it < 4; ++it) {
      int idx = tid + it * 256;        // 0..1023
      int row = idx >> 3;              // 0..127
      int chv = (idx & 7) << 2;        // 0,4,...,28
      v4f xa = *(const v4f*)&X[(size_t)(mBase + row) * CH + k0 + chv];
      v4f wa = *(const v4f*)&W[(size_t)(nBase + row) * CH + k0 + chv];
      v4bf xb, wb;
      for (int e = 0; e < 4; ++e) { xb[e] = (bf16_t)xa[e]; wb[e] = (bf16_t)wa[e]; }
      *(v4bf*)&As[row * 40 + chv] = xb;
      *(v4bf*)&Bs[row * 40 + chv] = wb;
    }
    __syncthreads();
    if (k0 + 32 < CH)   // pull next A slab toward L2/WGP$ (global_prefetch_b8)
      __builtin_prefetch(&X[(size_t)(mBase + (tid >> 1)) * CH + k0 + 32 + ((tid & 1) << 4)], 0, 1);
    v16bf bfrag[2];
    for (int ni = 0; ni < 2; ++ni)
      bfrag[ni] = load_fragB(Bs, wn * 32 + ni * 16, 40, 0, lane);
    for (int mi = 0; mi < 4; ++mi) {
      v16bf afrag = load_fragA(As, wm * 64 + mi * 16, 40, 0, lane);
      for (int ni = 0; ni < 2; ++ni)
        acc[mi][ni] = wmma_bf16f32(afrag, bfrag[ni], acc[mi][ni]);
    }
  }

  const int nLane = lane & 15;
  const int hl    = lane >> 4;
  for (int ni = 0; ni < 2; ++ni) {
    int c = nBase + wn * 32 + ni * 16 + nLane;
    float bval = bias[c];
    for (int mi = 0; mi < 4; ++mi) {
      int m0 = mBase + wm * 64 + mi * 16 + hl * 8;   // D: lane holds col c, vgprs m0..m0+7
      if (which < 2) {
        bf16_t* Out = (which == 0) ? Qbf : Kbf;
        for (int v = 0; v < 8; ++v)
          Out[(size_t)(m0 + v) * CH + c] = (bf16_t)(acc[mi][ni][v] + bval);
      } else {
        int b = m0 >> 12, tok = m0 & 4095;           // tiles never straddle batches
        v8bf pk;
        for (int v = 0; v < 8; ++v) pk[v] = (bf16_t)(acc[mi][ni][v] + bval);
        *(v8bf*)&Vt[((size_t)b * CH + c) * NTOK + tok] = pk;   // contiguous along tok
      }
    }
  }
}

// ---------------------------------------------------------------------------
// Kernel 3: agent pooling (AdaptiveAvgPool1d over 256-token chunks of Q),
// scale = hd^-0.5 folded in.  agent[b][h][a][d] bf16.
// ---------------------------------------------------------------------------
__global__ __launch_bounds__(768) void pool_kernel(const bf16_t* __restrict__ Qbf,
                                                   bf16_t* __restrict__ agent) {
  int b = blockIdx.x >> 4, a = blockIdx.x & 15;
  int c = threadIdx.x;
  const bf16_t* p = Qbf + (size_t)(b * NTOK + a * 256) * CH + c;
  float s = 0.f;
  for (int t = 0; t < 256; ++t) s += (float)p[(size_t)t * CH];
  int h = c >> 6, d = c & 63;
  agent[(((size_t)b * NHEAD + h) * NAGENT + a) * HDIM + d] = (bf16_t)(s * (0.125f / 256.f));
}

// ---------------------------------------------------------------------------
// Kernel 4: stage-1 logits  L[bh][a][m] = (scale*ah) @ kh^T   (WMMA, K=64)
// ---------------------------------------------------------------------------
__global__ __launch_bounds__(256) void s1_logits_kernel(const bf16_t* __restrict__ agent,
                                                        const bf16_t* __restrict__ Kbf,
                                                        float* __restrict__ Lbuf) {
  int bh = blockIdx.y, b = bh / NHEAD, h = bh % NHEAD;
  int mBase = blockIdx.x * 128;
  int lane = threadIdx.x & 31, wave = threadIdx.x >> 5;
  const bf16_t* ag = agent + (size_t)bh * NAGENT * HDIM;           // [16][64]
  const bf16_t* Kh = Kbf + (size_t)b * NTOK * CH + h * HDIM;       // rows pitch CH
  v8f acc; for (int e = 0; e < 8; ++e) acc[e] = 0.f;
  for (int ks = 0; ks < 2; ++ks) {
    v16bf af = load_fragA(ag, 0, HDIM, ks * 32, lane);
    v16bf br = load_fragB(Kh, mBase + wave * 16, CH, ks * 32, lane);
    acc = wmma_bf16f32(af, br, acc);
  }
  int mcol = mBase + wave * 16 + (lane & 15);
  int a0 = (lane >> 4) * 8;
  float* Lp = Lbuf + (size_t)bh * NAGENT * NTOK + mcol;
  for (int v = 0; v < 8; ++v) Lp[(size_t)(a0 + v) * NTOK] = acc[v];
}

// ---------------------------------------------------------------------------
// Kernel 5: exact row softmax over m=4096, P bf16 (pb bias provably cancels).
// ---------------------------------------------------------------------------
__global__ __launch_bounds__(256) void s1_softmax_kernel(const float* __restrict__ Lbuf,
                                                         bf16_t* __restrict__ Pbuf) {
  __shared__ float red[256];
  int row = blockIdx.x;                         // bh*16 + a
  const float* L = Lbuf + (size_t)row * NTOK;
  bf16_t* P = Pbuf + (size_t)row * NTOK;
  int t = threadIdx.x;
  float mx = -3.4e38f;
  for (int i = t; i < NTOK; i += 256) mx = fmaxf(mx, L[i]);
  red[t] = mx; __syncthreads();
  for (int s = 128; s > 0; s >>= 1) { if (t < s) red[t] = fmaxf(red[t], red[t + s]); __syncthreads(); }
  mx = red[0]; __syncthreads();
  float sum = 0.f;
  for (int i = t; i < NTOK; i += 256) sum += __expf(L[i] - mx);
  red[t] = sum; __syncthreads();
  for (int s = 128; s > 0; s >>= 1) { if (t < s) red[t] += red[t + s]; __syncthreads(); }
  float inv = 1.f / red[0];
  for (int i = t; i < NTOK; i += 256) P[i] = (bf16_t)(__expf(L[i] - mx) * inv);
}

// ---------------------------------------------------------------------------
// Kernel 6: agent_v = P @ vh  via P[16][4096] x Vt[64][4096], 8-way K split,
// ds_add_f32 LDS reduction, stored transposed+zero-padded: agentvT[bh][d][32].
// ---------------------------------------------------------------------------
__global__ __launch_bounds__(256) void s1_av_kernel(const bf16_t* __restrict__ Pbuf,
                                                    const bf16_t* __restrict__ Vt,
                                                    bf16_t* __restrict__ agentvT) {
  int bh = blockIdx.x, b = bh / NHEAD, h = bh % NHEAD;
  int tid = threadIdx.x, lane = tid & 31, wave = tid >> 5;
  const bf16_t* P = Pbuf + (size_t)bh * NAGENT * NTOK;             // [16][4096]
  const bf16_t* V = Vt + ((size_t)b * CH + h * HDIM) * NTOK;       // [64][4096]
  v8f acc[4];
  for (int di = 0; di < 4; ++di) for (int e = 0; e < 8; ++e) acc[di][e] = 0.f;
  for (int ks = wave * 16; ks < wave * 16 + 16; ++ks) {            // 128 K-steps total
    int k0 = ks * 32;
    v16bf af = load_fragA(P, 0, NTOK, k0, lane);
    for (int di = 0; di < 4; ++di) {
      v16bf br = load_fragB(V, di * 16, NTOK, k0, lane);
      acc[di] = wmma_bf16f32(af, br, acc[di]);
    }
  }
  __shared__ float red[NAGENT * HDIM];
  for (int i = tid; i < NAGENT * HDIM; i += 256) red[i] = 0.f;
  __syncthreads();
  int dcol = lane & 15, a0 = (lane >> 4) * 8;
  for (int di = 0; di < 4; ++di)
    for (int v = 0; v < 8; ++v)
      atomicAdd(&red[(a0 + v) * HDIM + di * 16 + dcol], acc[di][v]);  // ds_add_f32
  __syncthreads();
  if (tid < HDIM) {
    bf16_t* o = agentvT + (size_t)bh * HDIM * 32 + tid * 32;
    for (int a = 0; a < 16; ++a) o[a] = (bf16_t)red[a * HDIM + tid];
    for (int a = 16; a < 32; ++a) o[a] = (bf16_t)0.f;                // pad K to 32
  }
}

// ---------------------------------------------------------------------------
// Kernel 7: fused stage-2.  logits = qh @ (scale*ah)^T + ab -> softmax over 16
// agents (shfl_xor within lane-halves) -> P staged in LDS as A fragment ->
// x = P(16x32) @ agent_v(32x64) WMMA -> fp32 output (b,n,C layout).
// ---------------------------------------------------------------------------
__global__ __launch_bounds__(256) void s2_kernel(const bf16_t* __restrict__ Qbf,
                                                 const bf16_t* __restrict__ agent,
                                                 const bf16_t* __restrict__ agentvT,
                                                 const float* __restrict__ ab,
                                                 float* __restrict__ out) {
  int bh = blockIdx.y, b = bh / NHEAD, h = bh % NHEAD;
  int mBase = blockIdx.x * 128;
  int tid = threadIdx.x, lane = tid & 31, wave = tid >> 5;
  const bf16_t* Qh = Qbf + (size_t)b * NTOK * CH + h * HDIM;
  const bf16_t* ag = agent + (size_t)bh * NAGENT * HDIM;
  int tokBase = mBase + wave * 16;
  v8f acc; for (int e = 0; e < 8; ++e) acc[e] = 0.f;
  for (int ks = 0; ks < 2; ++ks) {
    v16bf af = load_fragA(Qh, tokBase, CH, ks * 32, lane);
    v16bf br = load_fragB(ag, 0, HDIM, ks * 32, lane);
    acc = wmma_bf16f32(af, br, acc);
  }
  float abv = ab[h * NAGENT + (lane & 15)];     // per-lane column bias
  int aCol = lane & 15, hl = lane >> 4;
  float p[8];
  for (int v = 0; v < 8; ++v) {                 // softmax over 16 agents = 16 lanes
    float x = acc[v] + abv;
    float mx = x;
    for (int m = 1; m < 16; m <<= 1) mx = fmaxf(mx, __shfl_xor(mx, m, 32));
    float e = __expf(x - mx);
    float s = e;
    for (int m = 1; m < 16; m <<= 1) s += __shfl_xor(s, m, 32);
    p[v] = e / s;
  }
  __shared__ __align__(16) bf16_t Pl[8][16][32];   // per-wave P^T staging, a padded to 32
  for (int i = lane; i < 256; i += 32) Pl[wave][i >> 4][16 + (i & 15)] = (bf16_t)0.f;
  for (int v = 0; v < 8; ++v) Pl[wave][hl * 8 + v][aCol] = (bf16_t)p[v];
  __syncthreads();
  v16bf af = load_fragA(&Pl[wave][0][0], 0, 32, 0, lane);
  const bf16_t* avT = agentvT + (size_t)bh * HDIM * 32;
  for (int di = 0; di < 4; ++di) {
    v16bf br = load_fragB(avT, di * 16, 32, 0, lane);
    v8f z; for (int e = 0; e < 8; ++e) z[e] = 0.f;
    v8f x = wmma_bf16f32(af, br, z);
    int c = h * HDIM + di * 16 + (lane & 15);
    for (int v = 0; v < 8; ++v) {
      int tok = tokBase + hl * 8 + v;
      out[((size_t)b * NTOK + tok) * CH + c] = x[v];
    }
  }
}

// ---------------------------------------------------------------------------
extern "C" void kernel_launch(void* const* d_in, const int* in_sizes, int n_in,
                              void* d_out, int out_size, void* d_ws, size_t ws_size,
                              hipStream_t stream) {
  (void)in_sizes; (void)n_in; (void)out_size; (void)ws_size;
  const float* s1 = (const float*)d_in[0];
  const float* s2 = (const float*)d_in[1];
  const float* Wq = (const float*)d_in[2];
  const float* bq = (const float*)d_in[3];
  const float* Wk = (const float*)d_in[4];
  const float* bk = (const float*)d_in[5];
  const float* Wv = (const float*)d_in[6];
  const float* bv = (const float*)d_in[7];
  // d_in[8] an_bias, d_in[10] ah_bias, d_in[11] aw_bias: pb is softmax-invariant -> unused.
  const float* na_bias = (const float*)d_in[9];
  const float* ha_bias = (const float*)d_in[12];
  const float* wa_bias = (const float*)d_in[13];
  float* out = (float*)d_out;

  char* ws = (char*)d_ws;
  constexpr size_t SZ_Q   = (size_t)MTOT * CH * sizeof(bf16_t);         // 50.3 MB
  constexpr size_t SZ_K   = SZ_Q;
  constexpr size_t SZ_VT  = SZ_Q;
  constexpr size_t SZ_AG  = (size_t)BH * NAGENT * HDIM * sizeof(bf16_t);
  constexpr size_t SZ_AVT = (size_t)BH * HDIM * 32 * sizeof(bf16_t);
  constexpr size_t SZ_AB  = (size_t)NHEAD * NAGENT * sizeof(float);
  constexpr size_t SZ_L   = (size_t)BH * NAGENT * NTOK * sizeof(float); // 25.2 MB
  size_t off = 0;
  bf16_t* Qbf     = (bf16_t*)(ws + off); off += SZ_Q;
  bf16_t* Kbf     = (bf16_t*)(ws + off); off += SZ_K;
  bf16_t* Vt      = (bf16_t*)(ws + off); off += SZ_VT;
  bf16_t* agent   = (bf16_t*)(ws + off); off += SZ_AG;
  bf16_t* agentvT = (bf16_t*)(ws + off); off += SZ_AVT;
  float*  ab      = (float*) (ws + off); off += SZ_AB;
  float*  Lbuf    = (float*) (ws + off); off += SZ_L;
  bf16_t* Pbuf    = (bf16_t*)(ws + off);

  bias_kernel<<<1, 256, 0, stream>>>(na_bias, ha_bias, wa_bias, ab);
  qkv_gemm_kernel<<<dim3(MTOT / 128, CH / 128, 3), 256, 0, stream>>>(
      s1, s2, Wq, Wk, Wv, bq, bk, bv, Qbf, Kbf, Vt);
  pool_kernel<<<NB * NAGENT, CH, 0, stream>>>(Qbf, agent);
  s1_logits_kernel<<<dim3(NTOK / 128, BH), 256, 0, stream>>>(agent, Kbf, Lbuf);
  s1_softmax_kernel<<<BH * NAGENT, 256, 0, stream>>>(Lbuf, Pbuf);
  s1_av_kernel<<<BH, 256, 0, stream>>>(Pbuf, Vt, agentvT);
  s2_kernel<<<dim3(NTOK / 128, BH), 256, 0, stream>>>(Qbf, agent, agentvT, ab, out);
}
